// TransformerBlock_80109730005338
// MI455X (gfx1250) — matrix operand order repaired and checked by reading, not compiled, not run on hardware
//
#include <hip/hip_runtime.h>
#include <hip/hip_bf16.h>

typedef __attribute__((ext_vector_type(8)))  __bf16 bf16x8;
typedef __attribute__((ext_vector_type(16))) __bf16 bf16x16;
typedef __attribute__((ext_vector_type(8)))  float  floatx8;

__device__ __forceinline__ unsigned short f32_to_bf16(float f) {
    unsigned int u = __float_as_uint(f);
    unsigned int r = 0x7FFFu + ((u >> 16) & 1u);
    return (unsigned short)((u + r) >> 16);
}

// Load a 16-element bf16 fragment as two 16B chunks (hi chunk at +hi_off elements).
__device__ __forceinline__ bf16x16 ld_frag(const unsigned short* p, int hi_off) {
    bf16x8 lo = *(const bf16x8*)p;
    bf16x8 hi = *(const bf16x8*)(p + hi_off);
    return __builtin_shufflevector(lo, hi, 0,1,2,3,4,5,6,7,8,9,10,11,12,13,14,15);
}

// ---------------- LayerNorm: f32 in -> bf16 out, 1 block per row of 1024 ----
__global__ void ln_kernel(const float* __restrict__ x, const float* __restrict__ g,
                          const float* __restrict__ bb, unsigned short* __restrict__ out) {
    int row = blockIdx.x;
    int t = threadIdx.x;                       // 256 threads, 4 elems each
    const float4 v = ((const float4*)(x + (size_t)row * 1024))[t];
    float s  = v.x + v.y + v.z + v.w;
    float ss = v.x*v.x + v.y*v.y + v.z*v.z + v.w*v.w;
#pragma unroll
    for (int o = 16; o > 0; o >>= 1) { s += __shfl_xor(s, o); ss += __shfl_xor(ss, o); }
    __shared__ float red[2][8];
    if ((t & 31) == 0) { red[0][t >> 5] = s; red[1][t >> 5] = ss; }
    __syncthreads();
    float st = 0.f, sst = 0.f;
#pragma unroll
    for (int i = 0; i < 8; ++i) { st += red[0][i]; sst += red[1][i]; }
    float mu  = st * (1.0f / 1024.0f);
    float var = sst * (1.0f / 1024.0f) - mu * mu;
    float rs  = rsqrtf(var + 1e-5f);
    int c0 = t * 4;
    ushort4 o4;
    o4.x = f32_to_bf16((v.x - mu) * rs * g[c0+0] + bb[c0+0]);
    o4.y = f32_to_bf16((v.y - mu) * rs * g[c0+1] + bb[c0+1]);
    o4.z = f32_to_bf16((v.z - mu) * rs * g[c0+2] + bb[c0+2]);
    o4.w = f32_to_bf16((v.w - mu) * rs * g[c0+3] + bb[c0+3]);
    *(ushort4*)(out + (size_t)row * 1024 + c0) = o4;
}

// ------------- Weight transpose + f32->bf16: w[K][N] -> wT[N][K] ------------
__global__ void transpose_to_bf16(const float* __restrict__ w, unsigned short* __restrict__ wT,
                                  int K, int N) {
    __shared__ unsigned short tile[32][33];
    int k0 = blockIdx.y * 32, n0 = blockIdx.x * 32;
    int tx = threadIdx.x, ty = threadIdx.y;    // (32,8)
#pragma unroll
    for (int i = 0; i < 32; i += 8)
        tile[ty + i][tx] = f32_to_bf16(w[(size_t)(k0 + ty + i) * N + n0 + tx]);
    __syncthreads();
#pragma unroll
    for (int i = 0; i < 32; i += 8)
        wT[(size_t)(n0 + ty + i) * K + k0 + tx] = tile[tx][ty + i];
}

// ------------- V repack: qkv[b][n][2048+dg] -> vT[b][dg][n] (bf16) ----------
__global__ void transpose_v_kernel(const unsigned short* __restrict__ qkv,
                                   unsigned short* __restrict__ vT) {
    __shared__ unsigned short tile[32][33];
    int b = blockIdx.z;
    int n0 = blockIdx.x * 32, d0 = blockIdx.y * 32;
    const unsigned short* in = qkv + (size_t)b * 2048 * 3072 + 2048;
    unsigned short* out = vT + (size_t)b * 1024 * 2048;
    int tx = threadIdx.x, ty = threadIdx.y;
#pragma unroll
    for (int i = 0; i < 32; i += 8)
        tile[ty + i][tx] = in[(size_t)(n0 + ty + i) * 3072 + d0 + tx];
    __syncthreads();
#pragma unroll
    for (int i = 0; i < 32; i += 8)
        out[(size_t)(d0 + ty + i) * 2048 + n0 + tx] = tile[tx][ty + i];
}

// ---------------- Generic WMMA GEMM: C = epi(A @ BT^T) ----------------------
// A: bf16 [M][lda] row-major. BT: bf16 [N][K] row-major (i.e. B transposed).
// Block 128 thr = 4 waves; block tile 128x128; wave tile 64x64 (16 wmma tiles).
// K-loop is software pipelined: next 8 fragments prefetched during the WMMAs.
// __launch_bounds__(128,1): allow full VGPR budget so the pipeline never spills.
template<bool HAS_BIAS, bool GELU, bool HAS_RES, bool OUT_F32>
__global__ void __launch_bounds__(128, 1)
gemm_wmma(const unsigned short* __restrict__ A, int lda,
          const unsigned short* __restrict__ BT,
          int M, int N, int K,
          const float* __restrict__ bias,
          const float* __restrict__ res,
          float* __restrict__ outF,
          unsigned short* __restrict__ outB) {
    int tid = threadIdx.x;
    int wave = tid >> 5, lane = tid & 31;
    int lhalf = lane >> 4, lmod = lane & 15;
    int row0 = blockIdx.y * 128 + (wave >> 1) * 64;
    int col0 = blockIdx.x * 128 + (wave & 1) * 64;

    floatx8 acc[4][4] = {};
    const unsigned short* Ap = A + ((size_t)row0 + lmod) * lda + lhalf * 8;
    const unsigned short* Bp = BT + ((size_t)col0 + lmod) * K + lhalf * 8;

    bf16x16 aF[4], bF[4];
#pragma unroll
    for (int mi = 0; mi < 4; ++mi) aF[mi] = ld_frag(Ap + (size_t)mi * 16 * lda, 16);
#pragma unroll
    for (int ni = 0; ni < 4; ++ni) bF[ni] = ld_frag(Bp + (size_t)ni * 16 * K, 16);

    for (int k0 = 0; k0 < K - 32; k0 += 32) {
        // prefetch next k-step while current WMMAs run
        bf16x16 aN[4], bN[4];
#pragma unroll
        for (int mi = 0; mi < 4; ++mi)
            aN[mi] = ld_frag(Ap + (size_t)mi * 16 * lda + k0 + 32, 16);
#pragma unroll
        for (int ni = 0; ni < 4; ++ni)
            bN[ni] = ld_frag(Bp + (size_t)ni * 16 * K + k0 + 32, 16);
#pragma unroll
        for (int mi = 0; mi < 4; ++mi)
#pragma unroll
            for (int ni = 0; ni < 4; ++ni)
                acc[mi][ni] = __builtin_amdgcn_wmma_f32_16x16x32_bf16(
                    false, aF[mi], false, bF[ni], (short)0, acc[mi][ni], false, false);
#pragma unroll
        for (int mi = 0; mi < 4; ++mi) aF[mi] = aN[mi];
#pragma unroll
        for (int ni = 0; ni < 4; ++ni) bF[ni] = bN[ni];
    }
    // last k-step
#pragma unroll
    for (int mi = 0; mi < 4; ++mi)
#pragma unroll
        for (int ni = 0; ni < 4; ++ni)
            acc[mi][ni] = __builtin_amdgcn_wmma_f32_16x16x32_bf16(
                false, aF[mi], false, bF[ni], (short)0, acc[mi][ni], false, false);

#pragma unroll
    for (int mi = 0; mi < 4; ++mi)
#pragma unroll
        for (int ni = 0; ni < 4; ++ni)
#pragma unroll
            for (int r = 0; r < 8; ++r) {
                int row = row0 + mi * 16 + r + lhalf * 8;
                int col = col0 + ni * 16 + lmod;
                float v = acc[mi][ni][r];
                if (HAS_BIAS) v += bias[col];
                if (GELU)     v = 0.5f * v * (1.0f + erff(v * 0.70710678118f));
                if (HAS_RES)  v += res[(size_t)row * N + col];
                if (OUT_F32)  outF[(size_t)row * N + col] = v;
                else          outB[(size_t)row * N + col] = f32_to_bf16(v);
            }
}

// ---------------- Flash attention: per (b,h), 16 query rows per wave --------
// qkv bf16 [b*2048][3072] (q|k|v). vT bf16 [b][dg][n]. out bf16 [b*2048][1024].
// Per key-tile: batch-load 8 K frags -> 8 WMMAs; issue 8 V frag loads, then
// overlap them with the online-softmax VALU work and the LDS P round-trip.
// __launch_bounds__(256,1): full VGPR budget -> qF/acc stay resident (no spill).
__global__ void __launch_bounds__(256, 1)
attn_kernel(const unsigned short* __restrict__ qkv,
            const unsigned short* __restrict__ vT,
            unsigned short* __restrict__ attn) {
    __shared__ __align__(16) unsigned short pbuf[8][16 * 64];
    int tid = threadIdx.x;
    int wave = tid >> 5, lane = tid & 31;
    int lhalf = lane >> 4, lmod = lane & 15;
    int b = blockIdx.z, h = blockIdx.y;
    int q0 = blockIdx.x * 128 + wave * 16;

    const unsigned short* Q  = qkv + (size_t)b * 2048 * 3072 + (size_t)h * 64;
    const unsigned short* Km = qkv + (size_t)b * 2048 * 3072 + 1024 + (size_t)h * 64;
    const unsigned short* Vt = vT + ((size_t)b * 1024 + (size_t)h * 64) * 2048;

    bf16x16 qF[2];
#pragma unroll
    for (int kk = 0; kk < 2; ++kk)
        qF[kk] = ld_frag(Q + (size_t)(q0 + lmod) * 3072 + kk * 32 + lhalf * 8, 16);

    floatx8 acc[4] = {};
    float mrow[8], lrow[8];
#pragma unroll
    for (int r = 0; r < 8; ++r) { mrow[r] = -3.0e38f; lrow[r] = 0.0f; }

    const float scale = 1.0f / 64.0f;   // reference divides by dim_head

    for (int kt = 0; kt < 32; ++kt) {
        // ---- batch load all K fragments for this 64-key tile ----
        const unsigned short* Kb = Km + (size_t)(kt * 64 + lmod) * 3072 + lhalf * 8;
        bf16x16 bK[4][2];
#pragma unroll
        for (int ni = 0; ni < 4; ++ni)
#pragma unroll
            for (int kk = 0; kk < 2; ++kk)
                bK[ni][kk] = ld_frag(Kb + (size_t)ni * 16 * 3072 + kk * 32, 16);
        // ---- S = Q @ K^T ----
        floatx8 s[4];
#pragma unroll
        for (int ni = 0; ni < 4; ++ni) {
            floatx8 z = {};
            z = __builtin_amdgcn_wmma_f32_16x16x32_bf16(false, qF[0], false, bK[ni][0],
                                                        (short)0, z, false, false);
            z = __builtin_amdgcn_wmma_f32_16x16x32_bf16(false, qF[1], false, bK[ni][1],
                                                        (short)0, z, false, false);
            s[ni] = z * scale;
        }
        // ---- issue all V fragment loads now; softmax below hides them ----
        const unsigned short* Vb = Vt + (size_t)lmod * 2048 + kt * 64 + lhalf * 8;
        bf16x16 bV[4][2];
#pragma unroll
        for (int di = 0; di < 4; ++di)
#pragma unroll
            for (int kk = 0; kk < 2; ++kk)
                bV[di][kk] = ld_frag(Vb + (size_t)di * 16 * 2048 + kk * 32, 16);
        // ---- online softmax per row (rows r + 8*lhalf; xor<=8 stays in half) ----
#pragma unroll
        for (int r = 0; r < 8; ++r) {
            float v = fmaxf(fmaxf(s[0][r], s[1][r]), fmaxf(s[2][r], s[3][r]));
#pragma unroll
            for (int o = 1; o < 16; o <<= 1) v = fmaxf(v, __shfl_xor(v, o));
            float mn = fmaxf(mrow[r], v);
            float alpha = __expf(mrow[r] - mn);
            mrow[r] = mn;
            lrow[r] *= alpha;
#pragma unroll
            for (int di = 0; di < 4; ++di) acc[di][r] *= alpha;
            float sum = 0.0f;
#pragma unroll
            for (int ni = 0; ni < 4; ++ni) {
                float p = __expf(s[ni][r] - mn);
                s[ni][r] = p;
                sum += p;
            }
#pragma unroll
            for (int o = 1; o < 16; o <<= 1) sum += __shfl_xor(sum, o);
            lrow[r] += sum;
        }
        // ---- P: C-layout -> LDS -> A-layout fragments ----
        unsigned short* pw = pbuf[wave];
#pragma unroll
        for (int ni = 0; ni < 4; ++ni)
#pragma unroll
            for (int r = 0; r < 8; ++r)
                pw[(r + lhalf * 8) * 64 + ni * 16 + lmod] = f32_to_bf16(s[ni][r]);

        bf16x16 pF[2];
#pragma unroll
        for (int kk = 0; kk < 2; ++kk)
            pF[kk] = ld_frag(pw + lmod * 64 + kk * 32 + lhalf * 8, 16);

        // ---- O += P @ V ----
#pragma unroll
        for (int di = 0; di < 4; ++di) {
            acc[di] = __builtin_amdgcn_wmma_f32_16x16x32_bf16(false, pF[0], false, bV[di][0],
                                                              (short)0, acc[di], false, false);
            acc[di] = __builtin_amdgcn_wmma_f32_16x16x32_bf16(false, pF[1], false, bV[di][1],
                                                              (short)0, acc[di], false, false);
        }
    }
    float rl[8];
#pragma unroll
    for (int r = 0; r < 8; ++r) rl[r] = 1.0f / lrow[r];
#pragma unroll
    for (int di = 0; di < 4; ++di)
#pragma unroll
        for (int r = 0; r < 8; ++r) {
            int row = q0 + r + lhalf * 8;
            int col = h * 64 + di * 16 + lmod;
            attn[((size_t)b * 2048 + row) * 1024 + col] = f32_to_bf16(acc[di][r] * rl[r]);
        }
}

extern "C" void kernel_launch(void* const* d_in, const int* in_sizes, int n_in,
                              void* d_out, int out_size, void* d_ws, size_t ws_size,
                              hipStream_t stream) {
    const float* x     = (const float*)d_in[0];
    const float* ln1_g = (const float*)d_in[1];
    const float* ln1_b = (const float*)d_in[2];
    const float* w_qkv = (const float*)d_in[3];
    const float* w_out = (const float*)d_in[4];
    const float* b_out = (const float*)d_in[5];
    const float* ln2_g = (const float*)d_in[6];
    const float* ln2_b = (const float*)d_in[7];
    const float* w1    = (const float*)d_in[8];
    const float* b1    = (const float*)d_in[9];
    const float* w2    = (const float*)d_in[10];
    const float* b2    = (const float*)d_in[11];

    char* ws = (char*)d_ws;
    const size_t Mi = 1024 * 1024;
    unsigned short* h     = (unsigned short*)(ws + 0);        //  8 MiB  [4096][1024]
    unsigned short* qkv   = (unsigned short*)(ws + 8 * Mi);   // 24 MiB  [4096][3072]
    unsigned short* vT    = (unsigned short*)(ws + 32 * Mi);  //  8 MiB  [2][1024][2048]
    unsigned short* attnb = (unsigned short*)(ws + 40 * Mi);  //  8 MiB  [4096][1024]
    float*          x1    = (float*)         (ws + 48 * Mi);  // 16 MiB  [4096][1024]
    unsigned short* h2    = (unsigned short*)(ws + 64 * Mi);  //  8 MiB
    unsigned short* act   = (unsigned short*)(ws + 72 * Mi);  // 32 MiB  [4096][4096]
    unsigned short* wqkvT = (unsigned short*)(ws + 104 * Mi); //  6 MiB  [3072][1024]
    unsigned short* woutT = (unsigned short*)(ws + 110 * Mi); //  2 MiB  [1024][1024]
    unsigned short* w1T   = (unsigned short*)(ws + 112 * Mi); //  8 MiB  [4096][1024]
    unsigned short* w2T   = (unsigned short*)(ws + 120 * Mi); //  8 MiB  [1024][4096]

    dim3 tb(32, 8);
    transpose_to_bf16<<<dim3(3072/32, 1024/32), tb, 0, stream>>>(w_qkv, wqkvT, 1024, 3072);
    transpose_to_bf16<<<dim3(1024/32, 1024/32), tb, 0, stream>>>(w_out, woutT, 1024, 1024);
    transpose_to_bf16<<<dim3(4096/32, 1024/32), tb, 0, stream>>>(w1,    w1T,   1024, 4096);
    transpose_to_bf16<<<dim3(1024/32, 4096/32), tb, 0, stream>>>(w2,    w2T,   4096, 1024);

    // h = LN1(x)
    ln_kernel<<<4096, 256, 0, stream>>>(x, ln1_g, ln1_b, h);
    // qkv = h @ w_qkv (no bias)
    gemm_wmma<false,false,false,false><<<dim3(3072/128, 4096/128), 128, 0, stream>>>(
        h, 1024, wqkvT, 4096, 3072, 1024, nullptr, nullptr, nullptr, qkv);
    // vT = transpose of V slice
    transpose_v_kernel<<<dim3(64, 32, 2), tb, 0, stream>>>(qkv, vT);
    // attention
    attn_kernel<<<dim3(16, 16, 2), 256, 0, stream>>>(qkv, vT, attnb);
    // x1 = x + attn @ w_out + b_out
    gemm_wmma<true,false,true,true><<<dim3(1024/128, 4096/128), 128, 0, stream>>>(
        attnb, 1024, woutT, 4096, 1024, 1024, b_out, x, x1, nullptr);
    // h2 = LN2(x1)
    ln_kernel<<<4096, 256, 0, stream>>>(x1, ln2_g, ln2_b, h2);
    // act = gelu(h2 @ w1 + b1)
    gemm_wmma<true,true,false,false><<<dim3(4096/128, 4096/128), 128, 0, stream>>>(
        h2, 1024, w1T, 4096, 4096, 1024, b1, nullptr, nullptr, act);
    // out = x1 + act @ w2 + b2
    gemm_wmma<true,false,true,true><<<dim3(1024/128, 4096/128), 128, 0, stream>>>(
        act, 4096, w2T, 4096, 1024, 4096, b2, x1, (float*)d_out, nullptr);
}
